// GPT_66340064854135
// MI455X (gfx1250) — matrix operand order repaired and checked by reading, not compiled, not run on hardware
//
#include <hip/hip_runtime.h>
#include <stdint.h>

// ---------------------------------------------------------------------------
// GPT-2-small forward for MI455X (gfx1250, wave32, WMMA).
// GEMMs + attention use v_wmma_f32_16x16x32_bf16 (bf16 in, f32 acc).
// GEMM tiles stream into LDS with global_load_async_to_lds_b128 (ASYNCcnt),
// double-buffered so WMMA overlaps the next tile's HBM/L2 traffic.
// ---------------------------------------------------------------------------

#define Dmodel 768
#define Tseq   1024
#define Bbatch 4
#define Hheads 12
#define HDim   64
#define Ffn    3072
#define Lnum   6
#define Vvocab 32000
#define Mrows  (Bbatch * Tseq)   // 4096

typedef __attribute__((ext_vector_type(16))) __bf16 v16bf;
typedef __attribute__((ext_vector_type(8)))  __bf16 v8bf;
typedef __attribute__((ext_vector_type(8)))  float  v8f;

union Frag16 {            // one wave32 WMMA 16-bit operand fragment (8 VGPRs)
  v16bf    v;
  uint32_t u[8];
  __bf16   b[16];
};

__device__ __forceinline__ v8f wmma_bf16(const Frag16& a, const Frag16& b, v8f c) {
  return __builtin_amdgcn_wmma_f32_16x16x32_bf16(
      /*neg_a=*/false, a.v, /*neg_b=*/false, b.v,
      /*c_mod=*/(short)0, c, /*reuse_a=*/false, /*reuse_b=*/false);
}

// 16B async copy global->LDS (gfx1250 GVS form: lds VGPR, voffset VGPR, saddr)
__device__ __forceinline__ void async_cp16(uint32_t lds_addr, uint32_t goff,
                                           uint64_t gbase) {
  asm volatile("global_load_async_to_lds_b128 %0, %1, %2"
               :: "v"(lds_addr), "v"(goff), "s"(gbase) : "memory");
}
__device__ __forceinline__ void async_wait0() {
  asm volatile("s_wait_asynccnt 0" ::: "memory");
}
__device__ __forceinline__ uint32_t lds_addr32(const void* p) {
  return (uint32_t)(uintptr_t)p;   // hardware truncates flat LDS addr to [31:0]
}

// ---------------------------------------------------------------------------
// Embedding: x[b,t,:] = tok_emb[idx[b,t],:] + pos_emb[t,:]
// ---------------------------------------------------------------------------
__global__ void embed_kernel(const int* __restrict__ idx,
                             const float* __restrict__ tok,
                             const float* __restrict__ pos,
                             float* __restrict__ x) {
  size_t i = (size_t)blockIdx.x * 256 + threadIdx.x;   // grid sized exactly
  int d = (int)(i % Dmodel);
  size_t bt = i / Dmodel;
  int tpos = (int)(bt % Tseq);
  x[i] = tok[(size_t)idx[bt] * Dmodel + d] + pos[(size_t)tpos * Dmodel + d];
}

// ---------------------------------------------------------------------------
// Weight convert+transpose: W[K][N] f32 -> Wt[N][K] bf16 (so GEMM B-operand
// rows are contiguous along K and tile loads are 16B ops).
// ---------------------------------------------------------------------------
__global__ void convert_transpose(const float* __restrict__ W,
                                  __bf16* __restrict__ Wt,
                                  int Kdim, int Ndim) {
  size_t i = (size_t)blockIdx.x * 256 + threadIdx.x;
  size_t total = (size_t)Kdim * Ndim;
  if (i >= total) return;
  size_t n  = i / (size_t)Kdim;
  size_t kk = i - n * (size_t)Kdim;
  Wt[i] = (__bf16)W[kk * (size_t)Ndim + n];
}

// ---------------------------------------------------------------------------
// LayerNorm over D=768, one 256-thread block per row, bf16 output (GEMM A).
// ---------------------------------------------------------------------------
__global__ __launch_bounds__(256) void layernorm_kernel(
    const float* __restrict__ x, const float* __restrict__ w,
    const float* __restrict__ b, __bf16* __restrict__ out) {
  const int row = blockIdx.x, t = threadIdx.x;
  const float* xr = x + (size_t)row * Dmodel;
  float v0 = xr[t], v1 = xr[t + 256], v2 = xr[t + 512];
  __shared__ float red[256];
  red[t] = v0 + v1 + v2;
  __syncthreads();
  for (int s = 128; s > 0; s >>= 1) { if (t < s) red[t] += red[t + s]; __syncthreads(); }
  const float mean = red[0] * (1.0f / Dmodel);
  __syncthreads();
  const float d0 = v0 - mean, d1 = v1 - mean, d2 = v2 - mean;
  red[t] = d0 * d0 + d1 * d1 + d2 * d2;
  __syncthreads();
  for (int s = 128; s > 0; s >>= 1) { if (t < s) red[t] += red[t + s]; __syncthreads(); }
  const float rs = rsqrtf(red[0] * (1.0f / Dmodel) + 1e-5f);
  __bf16* o = out + (size_t)row * Dmodel;
  o[t]       = (__bf16)(d0 * rs * w[t]       + b[t]);
  o[t + 256] = (__bf16)(d1 * rs * w[t + 256] + b[t + 256]);
  o[t + 512] = (__bf16)(d2 * rs * w[t + 512] + b[t + 512]);
}

// ---------------------------------------------------------------------------
// bf16 WMMA GEMM: out = act(A[M,K] @ Bt[N,K]^T + bias) (+ resid)
// 256 threads = 8 waves; 128x64 macro-tile; each wave owns 32x32 (4 WMMA
// accumulators, 2 A-frags x 2 B-frags per K-step of 32).
// LDS double-buffered; tiles streamed via global_load_async_to_lds_b128.
// Requires M%128==0, N%64==0, K%32==0 (holds for all call sites).
// ---------------------------------------------------------------------------
#define GBM 128
#define GBN 64
#define GBK 32
#define LDSP 48   // padded row stride (bf16) -> 96B rows, 16B aligned chunks

__global__ __launch_bounds__(256) void gemm_bf16(
    const __bf16* __restrict__ A, const __bf16* __restrict__ Bt,
    const float* __restrict__ bias, const float* __restrict__ resid,
    float* __restrict__ outF, __bf16* __restrict__ outB,
    int Mdim, int Ndim, int Kdim, int applyGelu) {
  __shared__ __bf16 As[2][GBM][LDSP];
  __shared__ __bf16 Bs[2][GBN][LDSP];
  const int t = threadIdx.x;
  const int lane = t & 31, wave = t >> 5;
  const int wm = wave & 3;                  // 32-row group 0..3
  const int wn = wave >> 2;                 // 32-col group 0..1
  const int m = lane & 15, half = lane >> 4;
  const int rowBase = blockIdx.y * GBM, colBase = blockIdx.x * GBN;
  const int lr = t >> 2, lc = (t & 3) * 8;  // cooperative 16B tile chunks

  const uint64_t Ab = (uint64_t)(uintptr_t)A;
  const uint64_t Bb = (uint64_t)(uintptr_t)Bt;
  (void)Mdim;

  auto prefetch = [&](int buf, int k0) {
    // A tile: 128 rows x 32 K -> 512 chunks, 2 per thread; B tile: 256 chunks.
    const uint32_t a0 = (uint32_t)(((size_t)(rowBase + lr) * Kdim + k0 + lc) * 2);
    const uint32_t a1 = (uint32_t)(((size_t)(rowBase + 64 + lr) * Kdim + k0 + lc) * 2);
    const uint32_t b0 = (uint32_t)(((size_t)(colBase + lr) * Kdim + k0 + lc) * 2);
    async_cp16(lds_addr32(&As[buf][lr][lc]),      a0, Ab);
    async_cp16(lds_addr32(&As[buf][64 + lr][lc]), a1, Ab);
    async_cp16(lds_addr32(&Bs[buf][lr][lc]),      b0, Bb);
  };

  v8f acc00 = {}, acc01 = {}, acc10 = {}, acc11 = {};
  prefetch(0, 0);
  async_wait0();
  __syncthreads();

  int cur = 0;
  for (int k0 = 0; k0 < Kdim; k0 += GBK) {
    const int nxt = cur ^ 1;
    if (k0 + GBK < Kdim) prefetch(nxt, k0 + GBK);   // overlap with WMMA below

    Frag16 a0, a1, b0, b1;
    // A fragments: lanes 0-15 K=0..15, lanes 16-31 K=16..31
    const uint32_t* ap0 = (const uint32_t*)&As[cur][wm * 32 + m][0];
    const uint32_t* ap1 = (const uint32_t*)&As[cur][wm * 32 + 16 + m][0];
#pragma unroll
    for (int i = 0; i < 4; ++i) {
      a0.u[i] = ap0[half * 8 + i];  a0.u[4 + i] = ap0[half * 8 + 4 + i];
      a1.u[i] = ap1[half * 8 + i];  a1.u[4 + i] = ap1[half * 8 + 4 + i];
    }
    // B fragments: lanes 0-15 K=0..15, lanes 16-31 K=16..31
    const uint32_t* bp0 = (const uint32_t*)&Bs[cur][wn * 32 + m][0];
    const uint32_t* bp1 = (const uint32_t*)&Bs[cur][wn * 32 + 16 + m][0];
#pragma unroll
    for (int i = 0; i < 8; ++i) { b0.u[i] = bp0[half * 8 + i]; b1.u[i] = bp1[half * 8 + i]; }

    acc00 = wmma_bf16(a0, b0, acc00);
    acc01 = wmma_bf16(a0, b1, acc01);
    acc10 = wmma_bf16(a1, b0, acc10);
    acc11 = wmma_bf16(a1, b1, acc11);

    async_wait0();     // my async writes to 'nxt' landed
    __syncthreads();   // everyone done reading 'cur' / writing 'nxt'
    cur = nxt;
  }

#pragma unroll
  for (int mi = 0; mi < 2; ++mi) {
#pragma unroll
    for (int r = 0; r < 8; ++r) {
      const int row = rowBase + wm * 32 + mi * 16 + r + 8 * half;
#pragma unroll
      for (int ni = 0; ni < 2; ++ni) {
        const int col = colBase + wn * 32 + ni * 16 + m;
        float v;
        if (mi == 0) v = (ni == 0) ? acc00[r] : acc01[r];
        else         v = (ni == 0) ? acc10[r] : acc11[r];
        if (bias)      v += bias[col];
        if (applyGelu) v = 0.5f * v * (1.0f + erff(v * 0.70710678118654752f));
        if (resid)     v += resid[(size_t)row * Ndim + col];
        if (outF) outF[(size_t)row * Ndim + col] = v;
        if (outB) outB[(size_t)row * Ndim + col] = (__bf16)v;
      }
    }
  }
}

// ---------------------------------------------------------------------------
// Flash attention: one wave per (16-query tile, batch, head).
// S = Q K^T via WMMA (HD=64 -> 2 K-steps), causal mask, online softmax,
// P staged through LDS into an A-fragment, O += P V via WMMA.
// ---------------------------------------------------------------------------
__global__ __launch_bounds__(32) void attn_flash(
    const __bf16* __restrict__ q, const __bf16* __restrict__ k,
    const __bf16* __restrict__ v, __bf16* __restrict__ y) {
  const int lane = threadIdx.x;
  const int qt = blockIdx.x;          // query tile 0..63
  const int bb = blockIdx.y;          // batch
  const int hh = blockIdx.z;          // head
  const int m = lane & 15, half = lane >> 4;
  const size_t headOff = (size_t)bb * Tseq * Dmodel + (size_t)hh * HDim;
  const __bf16* Qp = q + headOff;
  const __bf16* Kp = k + headOff;
  const __bf16* Vp = v + headOff;

  // Q as two 16x32 A-fragments (d = 0..31, 32..63); rows contiguous in d.
  Frag16 qa0, qa1;
  {
    const uint32_t* qr = (const uint32_t*)(Qp + (size_t)(qt * 16 + m) * Dmodel);
#pragma unroll
    for (int i = 0; i < 4; ++i) {
      qa0.u[i]     = qr[half * 8 + i];
      qa0.u[4 + i] = qr[half * 8 + 4 + i];
      qa1.u[i]     = qr[16 + half * 8 + i];
      qa1.u[4 + i] = qr[16 + half * 8 + 4 + i];
    }
  }

  float mrow[8], lrow[8];
  v8f o0 = {}, o1 = {}, o2 = {}, o3 = {};
#pragma unroll
  for (int r = 0; r < 8; ++r) { mrow[r] = -1e30f; lrow[r] = 0.0f; }
  const float scale = 0.125f;                // 1/sqrt(64)
  __shared__ __bf16 Plds[16][40];            // 16x32 P tile, padded stride
  const int nchunks = (qt + 2) >> 1;         // 32-key chunks up to causal edge

  for (int c = 0; c < nchunks; ++c) {
    const int kbase = c * 32;
    v8f s0 = {}, s1 = {};
    {   // S = Q K^T : B-fragment rows are K rows (contiguous in d)
      Frag16 kf;
      const uint32_t* kr0 = (const uint32_t*)(Kp + (size_t)(kbase + m) * Dmodel);
#pragma unroll
      for (int i = 0; i < 8; ++i) kf.u[i] = kr0[half * 8 + i];
      s0 = wmma_bf16(qa0, kf, s0);
#pragma unroll
      for (int i = 0; i < 8; ++i) kf.u[i] = kr0[16 + half * 8 + i];
      s0 = wmma_bf16(qa1, kf, s0);
      const uint32_t* kr1 = (const uint32_t*)(Kp + (size_t)(kbase + 16 + m) * Dmodel);
#pragma unroll
      for (int i = 0; i < 8; ++i) kf.u[i] = kr1[half * 8 + i];
      s1 = wmma_bf16(qa0, kf, s1);
#pragma unroll
      for (int i = 0; i < 8; ++i) kf.u[i] = kr1[16 + half * 8 + i];
      s1 = wmma_bf16(qa1, kf, s1);
    }

    // online softmax; row reductions stay within each 16-lane half
#pragma unroll
    for (int r = 0; r < 8; ++r) {
      const int qi = qt * 16 + r + 8 * half;
      float e0 = s0[r] * scale, e1 = s1[r] * scale;
      if (kbase + m > qi)      e0 = -1e30f;
      if (kbase + 16 + m > qi) e1 = -1e30f;
      float mx = fmaxf(e0, e1);
#pragma unroll
      for (int d = 1; d < 16; d <<= 1) mx = fmaxf(mx, __shfl_xor(mx, d, 32));
      const float mnew = fmaxf(mrow[r], mx);
      const float p0 = __expf(e0 - mnew), p1 = __expf(e1 - mnew);
      float ps = p0 + p1;
#pragma unroll
      for (int d = 1; d < 16; d <<= 1) ps += __shfl_xor(ps, d, 32);
      const float alpha = __expf(mrow[r] - mnew);
      lrow[r] = lrow[r] * alpha + ps;
      mrow[r] = mnew;
      o0[r] *= alpha; o1[r] *= alpha; o2[r] *= alpha; o3[r] *= alpha;
      Plds[r + 8 * half][m]      = (__bf16)p0;
      Plds[r + 8 * half][16 + m] = (__bf16)p1;
    }
    __syncthreads();   // forces dscnt wait before A-fragment gather

    Frag16 pf;   // P as 16x32 A-fragment
    {
      const uint32_t* pr = (const uint32_t*)&Plds[m][0];
#pragma unroll
      for (int i = 0; i < 4; ++i) { pf.u[i] = pr[half * 8 + i]; pf.u[4 + i] = pr[half * 8 + 4 + i]; }
    }
#pragma unroll
    for (int nt = 0; nt < 4; ++nt) {   // O[:, nt*16 .. nt*16+15] += P @ V
      Frag16 vf;
#pragma unroll
      for (int e = 0; e < 16; ++e)
        vf.b[e] = Vp[(size_t)(kbase + half * 16 + e) * Dmodel + nt * 16 + m];
      v8f& oo = (nt == 0) ? o0 : (nt == 1) ? o1 : (nt == 2) ? o2 : o3;
      oo = wmma_bf16(pf, vf, oo);
    }
    __syncthreads();   // Plds reused next chunk
  }

#pragma unroll
  for (int r = 0; r < 8; ++r) {
    const float inv = 1.0f / lrow[r];
    const size_t base = headOff + (size_t)(qt * 16 + r + 8 * half) * Dmodel;
    y[base +  0 + m] = (__bf16)(o0[r] * inv);
    y[base + 16 + m] = (__bf16)(o1[r] * inv);
    y[base + 32 + m] = (__bf16)(o2[r] * inv);
    y[base + 48 + m] = (__bf16)(o3[r] * inv);
  }
}

// ---------------------------------------------------------------------------
// Host orchestration
// ---------------------------------------------------------------------------
extern "C" void kernel_launch(void* const* d_in, const int* in_sizes, int n_in,
                              void* d_out, int out_size, void* d_ws, size_t ws_size,
                              hipStream_t stream) {
  (void)in_sizes; (void)n_in; (void)out_size; (void)ws_size;

  const int*   idx   = (const int*)  d_in[0];
  const float* tok   = (const float*)d_in[1];
  const float* pos   = (const float*)d_in[2];
  const float* ln1w  = (const float*)d_in[3];
  const float* ln1b  = (const float*)d_in[4];
  const float* wq    = (const float*)d_in[5];
  const float* bq    = (const float*)d_in[6];
  const float* wk    = (const float*)d_in[7];
  const float* bk    = (const float*)d_in[8];
  const float* wv    = (const float*)d_in[9];
  const float* bv    = (const float*)d_in[10];
  const float* wproj = (const float*)d_in[11];
  const float* bproj = (const float*)d_in[12];
  const float* ln2w  = (const float*)d_in[13];
  const float* ln2b  = (const float*)d_in[14];
  const float* w1    = (const float*)d_in[15];
  const float* b1    = (const float*)d_in[16];
  const float* w2    = (const float*)d_in[17];
  const float* b2    = (const float*)d_in[18];
  const float* lnfw  = (const float*)d_in[19];
  const float* lnfb  = (const float*)d_in[20];
  const float* headw = (const float*)d_in[21];

  // workspace carve-out (256B aligned slices)
  char* ws = (char*)d_ws;
  size_t off = 0;
  auto carve = [&](size_t bytes) {
    void* p = ws + off;
    off += (bytes + 255) & ~(size_t)255;
    return p;
  };
  float*  x  = (float*) carve((size_t)Mrows * Dmodel * 4);   // residual stream
  __bf16* h  = (__bf16*)carve((size_t)Mrows * Dmodel * 2);   // LN output
  __bf16* qB = (__bf16*)carve((size_t)Mrows * Dmodel * 2);
  __bf16* kB = (__bf16*)carve((size_t)Mrows * Dmodel * 2);
  __bf16* vB = (__bf16*)carve((size_t)Mrows * Dmodel * 2);
  __bf16* yB = (__bf16*)carve((size_t)Mrows * Dmodel * 2);   // attention out
  __bf16* a1 = (__bf16*)carve((size_t)Mrows * Ffn * 2);      // gelu(fc1)
  __bf16* wt = (__bf16*)carve((size_t)Dmodel * Vvocab * 2);  // staged bf16 weight

  const dim3 blk256(256);
  auto gemmGrid = [](int N, int M) { return dim3(N / GBN, M / GBM); };
  auto ctGrid = [](size_t total) { return dim3((unsigned)((total + 255) / 256)); };

  embed_kernel<<<dim3(Mrows * Dmodel / 256), blk256, 0, stream>>>(idx, tok, pos, x);

  for (int l = 0; l < Lnum; ++l) {
    const size_t wDD = (size_t)l * Dmodel * Dmodel;
    // ln1 -> h
    layernorm_kernel<<<dim3(Mrows), blk256, 0, stream>>>(x, ln1w + l * Dmodel, ln1b + l * Dmodel, h);
    // QKV projections (bf16 out for attention)
    convert_transpose<<<ctGrid((size_t)Dmodel * Dmodel), blk256, 0, stream>>>(wq + wDD, wt, Dmodel, Dmodel);
    gemm_bf16<<<gemmGrid(Dmodel, Mrows), blk256, 0, stream>>>(h, wt, bq + l * Dmodel, nullptr, nullptr, qB, Mrows, Dmodel, Dmodel, 0);
    convert_transpose<<<ctGrid((size_t)Dmodel * Dmodel), blk256, 0, stream>>>(wk + wDD, wt, Dmodel, Dmodel);
    gemm_bf16<<<gemmGrid(Dmodel, Mrows), blk256, 0, stream>>>(h, wt, bk + l * Dmodel, nullptr, nullptr, kB, Mrows, Dmodel, Dmodel, 0);
    convert_transpose<<<ctGrid((size_t)Dmodel * Dmodel), blk256, 0, stream>>>(wv + wDD, wt, Dmodel, Dmodel);
    gemm_bf16<<<gemmGrid(Dmodel, Mrows), blk256, 0, stream>>>(h, wt, bv + l * Dmodel, nullptr, nullptr, vB, Mrows, Dmodel, Dmodel, 0);
    // flash attention
    attn_flash<<<dim3(Tseq / 16, Bbatch, Hheads), dim3(32), 0, stream>>>(qB, kB, vB, yB);
    // output projection + residual (x updated in place, f32)
    convert_transpose<<<ctGrid((size_t)Dmodel * Dmodel), blk256, 0, stream>>>(wproj + wDD, wt, Dmodel, Dmodel);
    gemm_bf16<<<gemmGrid(Dmodel, Mrows), blk256, 0, stream>>>(yB, wt, bproj + l * Dmodel, x, x, nullptr, Mrows, Dmodel, Dmodel, 0);
    // ln2 -> h
    layernorm_kernel<<<dim3(Mrows), blk256, 0, stream>>>(x, ln2w + l * Dmodel, ln2b + l * Dmodel, h);
    // MLP: fc1+gelu (bf16 out), fc2 + residual (f32 out)
    convert_transpose<<<ctGrid((size_t)Dmodel * Ffn), blk256, 0, stream>>>(w1 + (size_t)l * Dmodel * Ffn, wt, Dmodel, Ffn);
    gemm_bf16<<<gemmGrid(Ffn, Mrows), blk256, 0, stream>>>(h, wt, b1 + l * Ffn, nullptr, nullptr, a1, Mrows, Ffn, Dmodel, 1);
    convert_transpose<<<ctGrid((size_t)Ffn * Dmodel), blk256, 0, stream>>>(w2 + (size_t)l * Ffn * Dmodel, wt, Ffn, Dmodel);
    gemm_bf16<<<gemmGrid(Dmodel, Mrows), blk256, 0, stream>>>(a1, wt, b2 + l * Dmodel, x, x, nullptr, Mrows, Dmodel, Ffn, 0);
  }

  // final LN + LM head (f32 logits straight into d_out)
  layernorm_kernel<<<dim3(Mrows), blk256, 0, stream>>>(x, lnfw, lnfb, h);
  convert_transpose<<<ctGrid((size_t)Dmodel * Vvocab), blk256, 0, stream>>>(headw, wt, Dmodel, Vvocab);
  gemm_bf16<<<gemmGrid(Vvocab, Mrows), blk256, 0, stream>>>(h, wt, nullptr, nullptr, (float*)d_out, nullptr, Mrows, Vvocab, Dmodel, 0);
}
